// AttentionLayer_11123965296955
// MI455X (gfx1250) — matrix operand order repaired and checked by reading, not compiled, not run on hardware
//
#include <hip/hip_runtime.h>

typedef __attribute__((ext_vector_type(16))) _Float16 v16h;
typedef __attribute__((ext_vector_type(8)))  _Float16 v8h;
typedef __attribute__((ext_vector_type(8)))  float    v8f;

// GCC-style vector matching the async-LDS builtin's parameter pointee type.
typedef int v4i_ __attribute__((vector_size(16)));
typedef __attribute__((address_space(1))) v4i_ gv4i;
typedef __attribute__((address_space(3))) v4i_ lv4i;

static constexpr int B  = 8;
static constexpr int T  = 4096;
static constexpr int D  = 64;
static constexpr int BC = 64;   // keys processed per iteration
static constexpr int NW = 4;    // waves per block
static constexpr int TQ = 16;   // query rows per wave
static constexpr int NCHUNK = T / BC;
static constexpr float LOG2E = 1.44269504088896340736f;

#if __has_builtin(__builtin_amdgcn_permlane16)
#define HAS_PERMLANE16 1
#endif
#if __has_builtin(__builtin_amdgcn_global_load_async_to_lds_b128) && \
    __has_builtin(__builtin_amdgcn_s_wait_asynccnt)
#define HAS_ASYNC_LDS 1
#endif

// ---- butterfly reductions across each 16-lane half of the wave ------------
__device__ inline float redmax16(float v) {
#ifdef HAS_PERMLANE16
#define RSTEP(s0, s1)                                                         \
    do {                                                                      \
        unsigned u = __float_as_uint(v);                                      \
        unsigned o = __builtin_amdgcn_permlane16(u, u, (s0), (s1), false, false); \
        v = fmaxf(v, __uint_as_float(o));                                     \
    } while (0)
    RSTEP(0x67452301u, 0xEFCDAB89u);   // xor 1
    RSTEP(0x54761032u, 0xDCFE98BAu);   // xor 2
    RSTEP(0x32107654u, 0xBA98FEDCu);   // xor 4
    RSTEP(0xFEDCBA98u, 0x76543210u);   // xor 8
#undef RSTEP
#else
    v = fmaxf(v, __shfl_xor(v, 1, 16));
    v = fmaxf(v, __shfl_xor(v, 2, 16));
    v = fmaxf(v, __shfl_xor(v, 4, 16));
    v = fmaxf(v, __shfl_xor(v, 8, 16));
#endif
    return v;
}

__device__ inline float redsum16(float v) {
#ifdef HAS_PERMLANE16
#define RSTEP(s0, s1)                                                         \
    do {                                                                      \
        unsigned u = __float_as_uint(v);                                      \
        unsigned o = __builtin_amdgcn_permlane16(u, u, (s0), (s1), false, false); \
        v = v + __uint_as_float(o);                                           \
    } while (0)
    RSTEP(0x67452301u, 0xEFCDAB89u);
    RSTEP(0x54761032u, 0xDCFE98BAu);
    RSTEP(0x32107654u, 0xBA98FEDCu);
    RSTEP(0xFEDCBA98u, 0x76543210u);
#undef RSTEP
#else
    v += __shfl_xor(v, 1, 16);
    v += __shfl_xor(v, 2, 16);
    v += __shfl_xor(v, 4, 16);
    v += __shfl_xor(v, 8, 16);
#endif
    return v;
}

// ---------------------------------------------------------------------------
// Kernel 1: Qh = f16((X @ W) * log2e), Xh = f16(X), XT = f16(X)^T per batch.
// ---------------------------------------------------------------------------
__global__ void __launch_bounds__(256)
attn_proj_kernel(const float* __restrict__ X, const float* __restrict__ W,
                 _Float16* __restrict__ Qh, _Float16* __restrict__ Xh,
                 _Float16* __restrict__ XT)
{
    __shared__ float Ws[D * D];
    const int tid = threadIdx.x;
    for (int i = tid; i < D * D; i += 256) Ws[i] = W[i];
    __syncthreads();

    const int row0 = blockIdx.x * 16;
    const int r    = row0 + (tid >> 4);
    const int e    = (tid & 15) * 4;
    const float* xr = X + (size_t)r * D;

    float a0 = 0.f, a1 = 0.f, a2 = 0.f, a3 = 0.f;
#pragma unroll 8
    for (int d = 0; d < D; ++d) {
        const float xv = xr[d];
        const float* wr = Ws + d * D + e;
        a0 += xv * wr[0]; a1 += xv * wr[1]; a2 += xv * wr[2]; a3 += xv * wr[3];
    }
    _Float16* qo = Qh + (size_t)r * D + e;
    qo[0] = (_Float16)(a0 * LOG2E); qo[1] = (_Float16)(a1 * LOG2E);
    qo[2] = (_Float16)(a2 * LOG2E); qo[3] = (_Float16)(a3 * LOG2E);

    const int b0 = row0 >> 12;
    const int t0 = row0 & (T - 1);
    const size_t base = (size_t)row0 * D;
    for (int i = tid; i < 16 * D; i += 256) {
        const _Float16 h = (_Float16)X[base + i];
        Xh[base + i] = h;
        const int rr = i >> 6, d = i & 63;
        XT[((size_t)b0 * D + d) * T + t0 + rr] = h;
    }
}

// ---------------------------------------------------------------------------
// Kernel 2: flash attention, f16 WMMA, async double-buffered staging.
// ---------------------------------------------------------------------------
__device__ inline v8f wmma_f16(v16h a, v16h b, v8f c) {
    return __builtin_amdgcn_wmma_f32_16x16x32_f16(false, a, false, b,
                                                  (short)0, c, false, false);
}

// stage one 64-key chunk: K row-major (64x64) + V^T d-major (64x64)
__device__ inline void stage_chunk(const uint4* __restrict__ K4,
                                   const uint4* __restrict__ V4,
                                   int kc, uint4* dK, uint4* dV, int tid)
{
#pragma unroll
    for (int k = 0; k < 4; ++k) {
        const int j = tid + k * 128;                     // 512 uint4 per tile
#ifdef HAS_ASYNC_LDS
        __builtin_amdgcn_global_load_async_to_lds_b128(
            (gv4i*)(K4 + (size_t)kc * 512 + j), (lv4i*)(dK + j), 0, 0);
#else
        dK[j] = K4[(size_t)kc * 512 + j];
#endif
    }
#pragma unroll
    for (int k = 0; k < 4; ++k) {
        const int j = tid + k * 128;
        const int d = j >> 3, part = j & 7;              // 8 uint4 per d-row
        const size_t src = (size_t)d * (T / 8) + (size_t)kc * 8 + part;
#ifdef HAS_ASYNC_LDS
        __builtin_amdgcn_global_load_async_to_lds_b128(
            (gv4i*)(V4 + src), (lv4i*)(dV + j), 0, 0);
#else
        dV[j] = V4[src];
#endif
    }
}

__global__ void __launch_bounds__(128)
attn_flash_kernel(const _Float16* __restrict__ Qh,
                  const _Float16* __restrict__ Xh,
                  const _Float16* __restrict__ XT,
                  float* __restrict__ Out)
{
    __shared__ __align__(16) _Float16 ldsK [2][BC * D];     // [buf][key][d]
    __shared__ __align__(16) _Float16 ldsVT[2][D * BC];     // [buf][d][key]
    __shared__ __align__(16) _Float16 ldsP [NW * TQ * BC];  // per-wave P scratch

    const int tid  = threadIdx.x;
    const int wave = tid >> 5;
    const int lane = tid & 31;
    const int lo16 = (lane < 16);
    const int l15  = lane & 15;

    const int bidx  = blockIdx.y;
    const int qrow0 = blockIdx.x * (NW * TQ) + wave * TQ;

    const uint4* K4 = (const uint4*)(Xh + (size_t)bidx * T * D);
    const uint4* V4 = (const uint4*)(XT + (size_t)bidx * D * T);

    const _Float16* qr = Qh + ((size_t)bidx * T + qrow0 + l15) * D;

    // ---- Q tile as two A-matrices (contiguous k = 16h + s order) -----
    const int klo = lo16 ? 0 : 16;
    v16h aQ0, aQ1;
    {
        v8h x0 = *(const v8h*)(qr + klo);
        v8h x1 = *(const v8h*)(qr + klo + 8);
        v8h x2 = *(const v8h*)(qr + 32 + klo);
        v8h x3 = *(const v8h*)(qr + 32 + klo + 8);
#pragma unroll
        for (int i = 0; i < 8; ++i) {
            aQ0[i] = x0[i]; aQ0[i + 8] = x1[i];
            aQ1[i] = x2[i]; aQ1[i + 8] = x3[i];
        }
    }

    float mrow[8], lrow[8];
    v8f o0 = {}, o1 = {}, o2 = {}, o3 = {};
#pragma unroll
    for (int r2 = 0; r2 < 8; ++r2) { mrow[r2] = -3.0e38f; lrow[r2] = 0.f; }

    _Float16* pbuf = ldsP + wave * (TQ * BC);

    // prologue: stage chunk 0 into buffer 0
    stage_chunk(K4, V4, 0, (uint4*)ldsK[0], (uint4*)ldsVT[0], tid);

    for (int kc = 0; kc < NCHUNK; ++kc) {
        const int cur = kc & 1;
#ifdef HAS_ASYNC_LDS
        __builtin_amdgcn_s_wait_asynccnt(0);   // our chunk-kc loads are in LDS
#endif
        __syncthreads();                        // all waves staged & done with buf[1-cur]
        if (kc + 1 < NCHUNK)                    // overlap next chunk's DMA with compute
            stage_chunk(K4, V4, kc + 1, (uint4*)ldsK[cur ^ 1], (uint4*)ldsVT[cur ^ 1], tid);

        const _Float16* Kc = ldsK[cur];
        const _Float16* Vc = ldsVT[cur];

        // ---- S = Q * K^T : four 16x16 f32 tiles (key groups of 16) -------
        const int bklo = lo16 ? 0 : 16;
        v8f s0 = {}, s1 = {}, s2 = {}, s3 = {};
        {
            const _Float16* k0 = Kc + (size_t)(0 * 16 + l15) * D + bklo;
            const _Float16* k1 = Kc + (size_t)(1 * 16 + l15) * D + bklo;
            const _Float16* k2 = Kc + (size_t)(2 * 16 + l15) * D + bklo;
            const _Float16* k3 = Kc + (size_t)(3 * 16 + l15) * D + bklo;
            s0 = wmma_f16(aQ0, *(const v16h*)(k0),      s0);
            s0 = wmma_f16(aQ1, *(const v16h*)(k0 + 32), s0);
            s1 = wmma_f16(aQ0, *(const v16h*)(k1),      s1);
            s1 = wmma_f16(aQ1, *(const v16h*)(k1 + 32), s1);
            s2 = wmma_f16(aQ0, *(const v16h*)(k2),      s2);
            s2 = wmma_f16(aQ1, *(const v16h*)(k2 + 32), s2);
            s3 = wmma_f16(aQ0, *(const v16h*)(k3),      s3);
            s3 = wmma_f16(aQ1, *(const v16h*)(k3 + 32), s3);
        }

        // ---- online softmax (log2 domain, deferred row-sum) --------------
#pragma unroll
        for (int r2 = 0; r2 < 8; ++r2) {
            float mv = fmaxf(fmaxf(s0[r2], s1[r2]), fmaxf(s2[r2], s3[r2]));
            mv = redmax16(mv);
            const float mnew = fmaxf(mrow[r2], mv);
            const float corr = exp2f(mrow[r2] - mnew);
            mrow[r2] = mnew;
            const float p0 = exp2f(s0[r2] - mnew);
            const float p1 = exp2f(s1[r2] - mnew);
            const float p2 = exp2f(s2[r2] - mnew);
            const float p3 = exp2f(s3[r2] - mnew);
            lrow[r2] = lrow[r2] * corr + ((p0 + p1) + (p2 + p3));
            o0[r2] *= corr; o1[r2] *= corr; o2[r2] *= corr; o3[r2] *= corr;
            const int prow = lo16 ? r2 : (8 + r2);
            _Float16* pw = pbuf + prow * BC + l15;
            pw[0]  = (_Float16)p0;
            pw[16] = (_Float16)p1;
            pw[32] = (_Float16)p2;
            pw[48] = (_Float16)p3;
        }

        // ---- re-read P as two A-matrices (16x64 over two k-steps) --------
        const int alo = lo16 ? 0 : 16;
        v16h aP0, aP1;
        {
            const _Float16* pr = pbuf + l15 * BC;
            v8h p0 = *(const v8h*)(pr + alo);
            v8h p1 = *(const v8h*)(pr + alo + 8);
            v8h p2 = *(const v8h*)(pr + 32 + alo);
            v8h p3 = *(const v8h*)(pr + 32 + alo + 8);
#pragma unroll
            for (int i = 0; i < 8; ++i) {
                aP0[i] = p0[i]; aP0[i + 8] = p1[i];
                aP1[i] = p2[i]; aP1[i + 8] = p3[i];
            }
        }

        // ---- O += P * V  (V^T d-major; keys 0-31 then 32-63) -------------
        const int vklo = lo16 ? 0 : 16;
        {
            const _Float16* v0 = Vc + (size_t)(0 * 16 + l15) * BC + vklo;
            const _Float16* v1 = Vc + (size_t)(1 * 16 + l15) * BC + vklo;
            const _Float16* v2 = Vc + (size_t)(2 * 16 + l15) * BC + vklo;
            const _Float16* v3 = Vc + (size_t)(3 * 16 + l15) * BC + vklo;
            o0 = wmma_f16(aP0, *(const v16h*)(v0),      o0);
            o0 = wmma_f16(aP1, *(const v16h*)(v0 + 32), o0);
            o1 = wmma_f16(aP0, *(const v16h*)(v1),      o1);
            o1 = wmma_f16(aP1, *(const v16h*)(v1 + 32), o1);
            o2 = wmma_f16(aP0, *(const v16h*)(v2),      o2);
            o2 = wmma_f16(aP1, *(const v16h*)(v2 + 32), o2);
            o3 = wmma_f16(aP0, *(const v16h*)(v3),      o3);
            o3 = wmma_f16(aP1, *(const v16h*)(v3 + 32), o3);
        }
    }

    // ---- epilogue: reduce l across the row, normalize, store -------------
#pragma unroll
    for (int r2 = 0; r2 < 8; ++r2) {
        const float lsum = redsum16(lrow[r2]);
        const float inv  = 1.0f / lsum;
        const int   prow = lo16 ? r2 : (8 + r2);
        float* orow = Out + ((size_t)bidx * T + qrow0 + prow) * D + l15;
        orow[0]  = o0[r2] * inv;
        orow[16] = o1[r2] * inv;
        orow[32] = o2[r2] * inv;
        orow[48] = o3[r2] * inv;
    }
}

// ---------------------------------------------------------------------------
extern "C" void kernel_launch(void* const* d_in, const int* in_sizes, int n_in,
                              void* d_out, int out_size, void* d_ws, size_t ws_size,
                              hipStream_t stream)
{
    const float* X = (const float*)d_in[0];   // [B,T,D] fp32
    const float* W = (const float*)d_in[1];   // [D,D]   fp32
    float* out = (float*)d_out;               // [B,T,D] fp32

    _Float16* Qh = (_Float16*)d_ws;                         // 4 MB
    _Float16* Xh = Qh + (size_t)B * T * D;                  // 4 MB
    _Float16* XT = Xh + (size_t)B * T * D;                  // 4 MB (transposed)

    attn_proj_kernel<<<(B * T) / 16, 256, 0, stream>>>(X, W, Qh, Xh, XT);

    dim3 grid(T / (NW * TQ), B);   // 64 x 8 blocks, 128 threads (4 waves)
    attn_flash_kernel<<<grid, 128, 0, stream>>>(Qh, Xh, XT, out);
}
